// MPNNModel_58059367907339
// MI455X (gfx1250) — matrix operand order repaired and checked by reading, not compiled, not run on hardware
//
#include <hip/hip_runtime.h>
#include <hip/hip_bf16.h>

#define N_NODES 100000
#define N_EDGES 1200000
#define HID 64
#define EDGE_DIM 16
#define N_LAYERS 3

#define EDGE_TILES (N_EDGES / 16)   // 75000
#define NODE_TILES (N_NODES / 16)   // 6250

typedef __attribute__((ext_vector_type(16))) __bf16 v16bf;
typedef __attribute__((ext_vector_type(8)))  __bf16 bf16x8;
typedef __attribute__((ext_vector_type(8)))  float  v8f;

__device__ __forceinline__ v8f wmma_bf16(v16bf a, v16bf b, v8f c) {
  return __builtin_amdgcn_wmma_f32_16x16x32_bf16(false, a, false, b, (short)0, c,
                                                 false, false);
}

// ---------------------------------------------------------------------------
// Fragment helpers.
// A-matrix (16x32 bf16, MxK): lane = half*16 + m (m=M), elements e:
//   e<8 : K = kt*32 + half*8 + e        (VGPR0-3)
//   e>=8: K = kt*32 + half*8 + 16 + e-8 (VGPR4-7)
// B-matrix (32x16 bf16, KxN): lane = half*16 + n (n=N), element e:
//   K = kt*32 + half*8 + e (e<8), kt*32 + half*8 + 16 + e-8 (e>=8)
// C/D (16x16 f32): lane = half*16 + n, VGPR j holds M = half*8 + j
// ---------------------------------------------------------------------------

// Load A-fragment from a contiguous bf16 feature row.
__device__ __forceinline__ v16bf a_frag_bf16(const __bf16* row, int kt, int hh) {
  union { v16bf v; bf16x8 h[2]; } u;
  u.h[0] = *(const bf16x8*)(row + kt * 32 + hh * 8);
  u.h[1] = *(const bf16x8*)(row + kt * 32 + hh * 8 + 16);
  return u.v;
}

// Load A-fragment from a contiguous fp32 feature row (convert to bf16).
__device__ __forceinline__ v16bf a_frag_f32(const float* row, int kt, int hh) {
  const float4* p = (const float4*)(row + kt * 32 + hh * 8);
  float4 x0 = p[0], x1 = p[1];
  const float4* q = (const float4*)(row + kt * 32 + hh * 8 + 16);
  float4 y0 = q[0], y1 = q[1];
  v16bf v;
  v[0] = (__bf16)x0.x; v[1] = (__bf16)x0.y; v[2] = (__bf16)x0.z; v[3] = (__bf16)x0.w;
  v[4] = (__bf16)x1.x; v[5] = (__bf16)x1.y; v[6] = (__bf16)x1.z; v[7] = (__bf16)x1.w;
  v[8] = (__bf16)y0.x; v[9] = (__bf16)y0.y; v[10] = (__bf16)y0.z; v[11] = (__bf16)y0.w;
  v[12] = (__bf16)y1.x; v[13] = (__bf16)y1.y; v[14] = (__bf16)y1.z; v[15] = (__bf16)y1.w;
  return v;
}

// Edge-feature A-fragment: 16 valid K columns, upper half zero-padded.
__device__ __forceinline__ v16bf a_frag_edge(const float* row, int hh) {
  const float4* p = (const float4*)(row + hh * 8);
  float4 x0 = p[0], x1 = p[1];
  v16bf v;
  v[0] = (__bf16)x0.x; v[1] = (__bf16)x0.y; v[2] = (__bf16)x0.z; v[3] = (__bf16)x0.w;
  v[4] = (__bf16)x1.x; v[5] = (__bf16)x1.y; v[6] = (__bf16)x1.z; v[7] = (__bf16)x1.w;
#pragma unroll
  for (int i = 8; i < 16; ++i) v[i] = (__bf16)0.0f;
  return v;
}

// Load a B-fragment (1024 bytes) from LDS.
__device__ __forceinline__ v16bf ldsfrag(const __bf16* base, int f, int lane) {
  return *(const v16bf*)(base + f * 512 + lane * 16);
}

// Convert fp32 weight matrix W[K,LD] (row-major) into bf16 B-fragments in LDS.
// Fragment f = kt*NT + nt; lane l; element e -> K = kt*32 + (e<8 ? (l>>4)*8 + e : 16 + (l>>4)*8 + e-8),
// N = nt*16 + (l&15). Rows K >= Krows are zero-filled (padding).
__device__ __forceinline__ void fill_bfrags(__bf16* dst, const float* W, int LD,
                                            int Krows, int KT, int NT,
                                            int tid, int nthreads) {
  int total = KT * NT * 512;
  for (int idx = tid; idx < total; idx += nthreads) {
    int f = idx >> 9;
    int r = idx & 511;
    int lane = r >> 4;
    int e = r & 15;
    int kt = f / NT, nt = f - kt * NT;
    int K = kt * 32 + (e < 8 ? (lane >> 4) * 8 + e : 16 + (lane >> 4) * 8 + (e - 8));
    int N = nt * 16 + (lane & 15);
    float v = (K < Krows) ? W[(size_t)K * LD + N] : 0.0f;
    dst[idx] = (__bf16)v;
  }
}

__device__ __forceinline__ float sigmoidf(float x) {
  return 1.0f / (1.0f + __expf(-x));
}

// ---------------------------------------------------------------------------
__global__ void zero_kernel(float* __restrict__ p, int n) {
  int n4 = n >> 2;
  float4 z = make_float4(0.f, 0.f, 0.f, 0.f);
  float4* p4 = (float4*)p;
  for (int i = blockIdx.x * blockDim.x + threadIdx.x; i < n4;
       i += gridDim.x * blockDim.x)
    p4[i] = z;
}

// ---------------------------------------------------------------------------
// Encoder: h = X @ enc_W + enc_b ; also writes bf16 shadow of h.
// ---------------------------------------------------------------------------
__global__ __launch_bounds__(256) void encoder_kernel(
    const float* __restrict__ X, const float* __restrict__ W,
    const float* __restrict__ b, float* __restrict__ h,
    __bf16* __restrict__ hbf) {
  __shared__ __align__(32) __bf16 sW[8 * 512];  // 2 kt x 4 nt
  __shared__ float sB[HID];
  int tid = threadIdx.x;
  fill_bfrags(sW, W, HID, HID, 2, 4, tid, 256);
  if (tid < HID) sB[tid] = b[tid];
  __syncthreads();

  int lane = tid & 31, wave = tid >> 5;
  int hh = lane >> 4, l15 = lane & 15;
  int nw = gridDim.x * 8;
  for (int tile = blockIdx.x * 8 + wave; tile < NODE_TILES; tile += nw) {
    int n0 = tile * 16;
    const float* row = X + (size_t)(n0 + l15) * HID;
    v16bf A0 = a_frag_f32(row, 0, hh);
    v16bf A1 = a_frag_f32(row, 1, hh);
#pragma unroll
    for (int nt = 0; nt < 4; ++nt) {
      v8f c = {};
      c = wmma_bf16(A0, ldsfrag(sW, 0 * 4 + nt, lane), c);
      c = wmma_bf16(A1, ldsfrag(sW, 1 * 4 + nt, lane), c);
      float bias = sB[nt * 16 + l15];
#pragma unroll
      for (int j = 0; j < 8; ++j) {
        float v = c[j] + bias;
        size_t idx = (size_t)(n0 + hh * 8 + j) * HID + nt * 16 + l15;
        h[idx] = v;
        hbf[idx] = (__bf16)v;
      }
    }
  }
}

// ---------------------------------------------------------------------------
// Edge kernel: for each 16-edge tile:
//   m = relu([h[src]|h[dst]|ef] @ W1 + b1) @ W2 + b2 ;  atomicAdd into agg[dst]
// ---------------------------------------------------------------------------
__global__ __launch_bounds__(256) void edge_msg_kernel(
    const __bf16* __restrict__ hbf, const int* __restrict__ edge_index,
    const float* __restrict__ ef, const float* __restrict__ W1,
    const float* __restrict__ b1, const float* __restrict__ W2,
    const float* __restrict__ b2, float* __restrict__ agg) {
  __shared__ __align__(32) __bf16 sW1[20 * 512];  // 5 kt x 4 nt (K padded 144->160)
  __shared__ __align__(32) __bf16 sW2[8 * 512];   // 2 kt x 4 nt
  __shared__ __align__(32) __bf16 sT[8][16 * HID];  // per-wave transpose scratch
  __shared__ float sB1[HID], sB2[HID];
  __shared__ int sDst[8][16];

  int tid = threadIdx.x;
  fill_bfrags(sW1, W1, HID, 2 * HID + EDGE_DIM, 5, 4, tid, 256);
  fill_bfrags(sW2, W2, HID, HID, 2, 4, tid, 256);
  if (tid < HID) { sB1[tid] = b1[tid]; sB2[tid] = b2[tid]; }
  __syncthreads();

  int lane = tid & 31, wave = tid >> 5;
  int hh = lane >> 4, l15 = lane & 15;
  int nw = gridDim.x * 8;
  for (int tile = blockIdx.x * 8 + wave; tile < EDGE_TILES; tile += nw) {
    int e0 = tile * 16;
    int esrc = edge_index[e0 + l15];
    int edst = edge_index[N_EDGES + e0 + l15];
    if (hh == 0) sDst[wave][l15] = edst;

    // Prefetch next tile's edge features while we compute.
    {
      int nxt = tile + nw;
      if (nxt < EDGE_TILES)
        __builtin_prefetch(ef + (size_t)nxt * 16 * EDGE_DIM + lane * 8, 0, 1);
    }

    const __bf16* rs = hbf + (size_t)esrc * HID;
    const __bf16* rd = hbf + (size_t)edst * HID;
    v16bf a0 = a_frag_bf16(rs, 0, hh);
    v16bf a1 = a_frag_bf16(rs, 1, hh);
    v16bf a2 = a_frag_bf16(rd, 0, hh);
    v16bf a3 = a_frag_bf16(rd, 1, hh);
    v16bf a4 = a_frag_edge(ef + (size_t)(e0 + l15) * EDGE_DIM, hh);

    // hidden = relu(cat @ W1 + b1)   -> sT (row-major bf16, 16x64)
#pragma unroll
    for (int nt = 0; nt < 4; ++nt) {
      v8f c = {};
      c = wmma_bf16(a0, ldsfrag(sW1, 0 * 4 + nt, lane), c);
      c = wmma_bf16(a1, ldsfrag(sW1, 1 * 4 + nt, lane), c);
      c = wmma_bf16(a2, ldsfrag(sW1, 2 * 4 + nt, lane), c);
      c = wmma_bf16(a3, ldsfrag(sW1, 3 * 4 + nt, lane), c);
      c = wmma_bf16(a4, ldsfrag(sW1, 4 * 4 + nt, lane), c);
      float bias = sB1[nt * 16 + l15];
#pragma unroll
      for (int j = 0; j < 8; ++j)
        sT[wave][(hh * 8 + j) * HID + nt * 16 + l15] =
            (__bf16)fmaxf(c[j] + bias, 0.0f);
    }
    // LDS ops are in-order within a wave; reload hidden as A-fragments.
    const __bf16* hrow = &sT[wave][l15 * HID];
    v16bf g0 = a_frag_bf16(hrow, 0, hh);
    v16bf g1 = a_frag_bf16(hrow, 1, hh);

    int dstv[8];
#pragma unroll
    for (int j = 0; j < 8; ++j) dstv[j] = sDst[wave][hh * 8 + j];

    // m = hidden @ W2 + b2 ; scatter-add
#pragma unroll
    for (int nt = 0; nt < 4; ++nt) {
      v8f c = {};
      c = wmma_bf16(g0, ldsfrag(sW2, 0 * 4 + nt, lane), c);
      c = wmma_bf16(g1, ldsfrag(sW2, 1 * 4 + nt, lane), c);
      float bias = sB2[nt * 16 + l15];
#pragma unroll
      for (int j = 0; j < 8; ++j)
        atomicAdd(&agg[(size_t)dstv[j] * HID + nt * 16 + l15], c[j] + bias);
    }
  }
}

// ---------------------------------------------------------------------------
// GRU: h = GRUCell(agg, h).  Gate order (r,z,n) in columns [0:64|64:128|128:192].
// Updates fp32 h and bf16 shadow in place (tile-private rows).
// ---------------------------------------------------------------------------
__global__ __launch_bounds__(256) void gru_kernel(
    const float* __restrict__ agg, float* __restrict__ h,
    __bf16* __restrict__ hbf, const float* __restrict__ Wih,
    const float* __restrict__ Whh, const float* __restrict__ bih,
    const float* __restrict__ bhh) {
  __shared__ __align__(32) __bf16 sWih[24 * 512];  // 2 kt x 12 nt
  __shared__ __align__(32) __bf16 sWhh[24 * 512];
  __shared__ float sBih[3 * HID], sBhh[3 * HID];
  int tid = threadIdx.x;
  fill_bfrags(sWih, Wih, 3 * HID, HID, 2, 12, tid, 256);
  fill_bfrags(sWhh, Whh, 3 * HID, HID, 2, 12, tid, 256);
  if (tid < 3 * HID) { sBih[tid] = bih[tid]; sBhh[tid] = bhh[tid]; }
  __syncthreads();

  int lane = tid & 31, wave = tid >> 5;
  int hh = lane >> 4, l15 = lane & 15;
  int nw = gridDim.x * 8;
  for (int tile = blockIdx.x * 8 + wave; tile < NODE_TILES; tile += nw) {
    int n0 = tile * 16;
    const float* arow = agg + (size_t)(n0 + l15) * HID;
    v16bf aA0 = a_frag_f32(arow, 0, hh);
    v16bf aA1 = a_frag_f32(arow, 1, hh);
    const __bf16* hrow = hbf + (size_t)(n0 + l15) * HID;
    v16bf hA0 = a_frag_bf16(hrow, 0, hh);
    v16bf hA1 = a_frag_bf16(hrow, 1, hh);

#pragma unroll
    for (int g = 0; g < 4; ++g) {  // 16-col groups of the hidden dim
      v8f gr = {}, gz = {}, gn = {}, hr = {}, hz = {}, hn = {};
      gr = wmma_bf16(aA0, ldsfrag(sWih, 0 * 12 + g, lane), gr);
      gr = wmma_bf16(aA1, ldsfrag(sWih, 1 * 12 + g, lane), gr);
      gz = wmma_bf16(aA0, ldsfrag(sWih, 0 * 12 + 4 + g, lane), gz);
      gz = wmma_bf16(aA1, ldsfrag(sWih, 1 * 12 + 4 + g, lane), gz);
      gn = wmma_bf16(aA0, ldsfrag(sWih, 0 * 12 + 8 + g, lane), gn);
      gn = wmma_bf16(aA1, ldsfrag(sWih, 1 * 12 + 8 + g, lane), gn);
      hr = wmma_bf16(hA0, ldsfrag(sWhh, 0 * 12 + g, lane), hr);
      hr = wmma_bf16(hA1, ldsfrag(sWhh, 1 * 12 + g, lane), hr);
      hz = wmma_bf16(hA0, ldsfrag(sWhh, 0 * 12 + 4 + g, lane), hz);
      hz = wmma_bf16(hA1, ldsfrag(sWhh, 1 * 12 + 4 + g, lane), hz);
      hn = wmma_bf16(hA0, ldsfrag(sWhh, 0 * 12 + 8 + g, lane), hn);
      hn = wmma_bf16(hA1, ldsfrag(sWhh, 1 * 12 + 8 + g, lane), hn);

      int col = g * 16 + l15;
      float bir = sBih[col], biz = sBih[HID + col], bin = sBih[2 * HID + col];
      float bhr = sBhh[col], bhz = sBhh[HID + col], bhn = sBhh[2 * HID + col];
#pragma unroll
      for (int j = 0; j < 8; ++j) {
        size_t idx = (size_t)(n0 + hh * 8 + j) * HID + col;
        float hv = h[idx];
        float r = sigmoidf(gr[j] + bir + hr[j] + bhr);
        float z = sigmoidf(gz[j] + biz + hz[j] + bhz);
        float n = tanhf(gn[j] + bin + r * (hn[j] + bhn));
        float out = (1.0f - z) * n + z * hv;
        h[idx] = out;
        hbf[idx] = (__bf16)out;
      }
    }
  }
}

// ---------------------------------------------------------------------------
// Readout: out = relu(h @ W1 + b1) @ W2 + b2
// ---------------------------------------------------------------------------
__global__ __launch_bounds__(256) void readout_kernel(
    const __bf16* __restrict__ hbf, const float* __restrict__ W1,
    const float* __restrict__ b1, const float* __restrict__ W2,
    const float* __restrict__ b2, float* __restrict__ out) {
  __shared__ __align__(32) __bf16 sW1[8 * 512];
  __shared__ __align__(32) __bf16 sW2[8 * 512];
  __shared__ __align__(32) __bf16 sT[8][16 * HID];
  __shared__ float sB1[HID], sB2[HID];
  int tid = threadIdx.x;
  fill_bfrags(sW1, W1, HID, HID, 2, 4, tid, 256);
  fill_bfrags(sW2, W2, HID, HID, 2, 4, tid, 256);
  if (tid < HID) { sB1[tid] = b1[tid]; sB2[tid] = b2[tid]; }
  __syncthreads();

  int lane = tid & 31, wave = tid >> 5;
  int hh = lane >> 4, l15 = lane & 15;
  int nw = gridDim.x * 8;
  for (int tile = blockIdx.x * 8 + wave; tile < NODE_TILES; tile += nw) {
    int n0 = tile * 16;
    const __bf16* row = hbf + (size_t)(n0 + l15) * HID;
    v16bf A0 = a_frag_bf16(row, 0, hh);
    v16bf A1 = a_frag_bf16(row, 1, hh);
#pragma unroll
    for (int nt = 0; nt < 4; ++nt) {
      v8f c = {};
      c = wmma_bf16(A0, ldsfrag(sW1, 0 * 4 + nt, lane), c);
      c = wmma_bf16(A1, ldsfrag(sW1, 1 * 4 + nt, lane), c);
      float bias = sB1[nt * 16 + l15];
#pragma unroll
      for (int j = 0; j < 8; ++j)
        sT[wave][(hh * 8 + j) * HID + nt * 16 + l15] =
            (__bf16)fmaxf(c[j] + bias, 0.0f);
    }
    const __bf16* trow = &sT[wave][l15 * HID];
    v16bf g0 = a_frag_bf16(trow, 0, hh);
    v16bf g1 = a_frag_bf16(trow, 1, hh);
#pragma unroll
    for (int nt = 0; nt < 4; ++nt) {
      v8f c = {};
      c = wmma_bf16(g0, ldsfrag(sW2, 0 * 4 + nt, lane), c);
      c = wmma_bf16(g1, ldsfrag(sW2, 1 * 4 + nt, lane), c);
      float bias = sB2[nt * 16 + l15];
#pragma unroll
      for (int j = 0; j < 8; ++j)
        out[(size_t)(n0 + hh * 8 + j) * HID + nt * 16 + l15] = c[j] + bias;
    }
  }
}

// ---------------------------------------------------------------------------
extern "C" void kernel_launch(void* const* d_in, const int* in_sizes, int n_in,
                              void* d_out, int out_size, void* d_ws,
                              size_t ws_size, hipStream_t stream) {
  (void)in_sizes; (void)n_in; (void)out_size; (void)ws_size;
  const float* node_features = (const float*)d_in[0];
  const int*   edge_index    = (const int*)d_in[1];
  const float* edge_features = (const float*)d_in[2];
  const float* enc_W  = (const float*)d_in[3];
  const float* enc_b  = (const float*)d_in[4];
  const float* msg_W1 = (const float*)d_in[5];
  const float* msg_b1 = (const float*)d_in[6];
  const float* msg_W2 = (const float*)d_in[7];
  const float* msg_b2 = (const float*)d_in[8];
  const float* gru_Wih = (const float*)d_in[9];
  const float* gru_Whh = (const float*)d_in[10];
  const float* gru_bih = (const float*)d_in[11];
  const float* gru_bhh = (const float*)d_in[12];
  const float* ro_W1 = (const float*)d_in[13];
  const float* ro_b1 = (const float*)d_in[14];
  const float* ro_W2 = (const float*)d_in[15];
  const float* ro_b2 = (const float*)d_in[16];
  float* out = (float*)d_out;

  // Workspace: h fp32 (25.6MB) | h bf16 (12.8MB) | agg fp32 (25.6MB)
  char* ws = (char*)d_ws;
  float*  h    = (float*)ws;
  __bf16* hbf  = (__bf16*)(ws + (size_t)N_NODES * HID * sizeof(float));
  float*  agg  = (float*)(ws + (size_t)N_NODES * HID * (sizeof(float) + sizeof(__bf16)));

  dim3 blk(256);
  encoder_kernel<<<782, blk, 0, stream>>>(node_features, enc_W, enc_b, h, hbf);
  for (int l = 0; l < N_LAYERS; ++l) {
    zero_kernel<<<2048, blk, 0, stream>>>(agg, N_NODES * HID);
    edge_msg_kernel<<<1024, blk, 0, stream>>>(
        hbf, edge_index, edge_features,
        msg_W1 + (size_t)l * (2 * HID + EDGE_DIM) * HID, msg_b1 + l * HID,
        msg_W2 + (size_t)l * HID * HID, msg_b2 + l * HID, agg);
    gru_kernel<<<782, blk, 0, stream>>>(
        agg, h, hbf, gru_Wih + (size_t)l * HID * 3 * HID,
        gru_Whh + (size_t)l * HID * 3 * HID, gru_bih + l * 3 * HID,
        gru_bhh + l * 3 * HID);
  }
  readout_kernel<<<782, blk, 0, stream>>>(hbf, ro_W1, ro_b1, ro_W2, ro_b2, out);
}
